// GRUDecoderGD_83691732730533
// MI455X (gfx1250) — matrix operand order repaired and checked by reading, not compiled, not run on hardware
//
#include <hip/hip_runtime.h>
#include <hip/hip_bf16.h>
#include <math.h>

// ---------------------------------------------------------------------------
// GRU decoder (2 layers, H=1024, B=64, V=32000, T=128) for gfx1250.
// All GEMMs run on v_wmma_f32_16x16x32_bf16 (wave32). Weights converted to
// bf16 once per call (they live in the 192MB L2 across all 128 steps);
// hidden state kept in fp32, rounded to bf16 only as WMMA input.
// A-panels staged to LDS in 32KB chunks via async global->LDS (ASYNCcnt).
// ---------------------------------------------------------------------------

typedef __bf16 bf16_t;
typedef __attribute__((ext_vector_type(16))) __bf16 v16bf;
typedef __attribute__((ext_vector_type(8)))  __bf16 v8bf;
typedef __attribute__((ext_vector_type(8)))  float  v8f;

#define Hs   1024
#define Bs   64
#define Vs   32000
#define Ts   128
#define Gs   (3 * Hs)
#define VTILES 500   // 32000 / 64
#define BK   256     // K-panel width staged in LDS (64 x 256 bf16 = 32 KB)

#if defined(__gfx1250__)
#define USE_ASYNC_LDS 1
#else
#define USE_ASYNC_LDS 0
#endif

__device__ __forceinline__ unsigned short f2bf(float x) {
  unsigned u = __builtin_bit_cast(unsigned, x);
  u = (u + 0x7FFFu + ((u >> 16) & 1u)) >> 16;
  return (unsigned short)u;
}

// ---- fragment builders (WMMA 16-bit operands, contiguous k order in A and B, wave32) -----------------
// A (16x32, MxK): lane<16 holds row M=lane, K in {ks..ks+15},
// ks = 0 for lanes 0-15, ks = 16 for lanes 16-31.  As is a [64][BK] LDS panel.
__device__ __forceinline__ v16bf make_afrag(const bf16_t* As, int mi, int lane,
                                            int kk) {
  int r  = mi * 16 + (lane & 15);
  int ks = (lane < 16) ? 0 : 16;
  v8bf lo = *(const v8bf*)(As + r * BK + kk + ks);
  v8bf hi = *(const v8bf*)(As + r * BK + kk + ks + 8);
  return __builtin_shufflevector(lo, hi, 0,1,2,3,4,5,6,7,8,9,10,11,12,13,14,15);
}

// B (32x16, KxN): lane<16 -> N=lane, K = k0..k0+15 ; lane>=16 -> N=lane-16,
// K = k0+16..k0+31. B[k][n] = W[n][k] (W row-major [N,1024]) -> one
// contiguous 32-byte load per lane.
__device__ __forceinline__ v16bf make_bfrag(const bf16_t* __restrict__ W,
                                            int n, int k0, int lane) {
  int kb = k0 + ((lane < 16) ? 0 : 16);
  return *(const v16bf*)(W + (size_t)n * Hs + kb);
}

// ---- stage one 64 x BK A-panel into LDS (async global->LDS, ASYNCcnt) ------
__device__ __forceinline__ void stage_chunk(const bf16_t* __restrict__ A,
                                            bf16_t* As, int ko) {
  const int tid = threadIdx.x;
#pragma unroll
  for (int i = 0; i < (Bs * BK) / (256 * 8); ++i) {   // 8 x b128 per thread
    int c   = i * 256 + tid;                          // 0..2047
    int row = c >> 5;                                 // BK/8 = 32 chunks/row
    int col = (c & 31) * 8;
    const bf16_t* gp = A + (size_t)row * Hs + ko * BK + col;
    bf16_t*       lp = As + row * BK + col;
#if USE_ASYNC_LDS
    // VDST = per-lane LDS byte offset (generic LDS ptr low 32 bits),
    // VADDR = 64-bit global address; tracked with ASYNCcnt.
    unsigned lds_off = (unsigned)(unsigned long long)lp;
    asm volatile("global_load_async_to_lds_b128 %0, %1, off"
                 :: "v"(lds_off), "v"(gp) : "memory");
#else
    *(v8bf*)lp = *(const v8bf*)gp;
#endif
  }
#if USE_ASYNC_LDS
  asm volatile("s_wait_asynccnt 0x0" ::: "memory");
#endif
}

// ---- shared GEMM core: C[64 x 64-tile] += A[64,1024] * W[Ntile,1024]^T -----
__device__ __forceinline__ void gemm_core(const bf16_t* __restrict__ A,
                                          const bf16_t* __restrict__ W,
                                          int Nbase, v8f acc[2], bf16_t* As) {
  const int tid  = threadIdx.x;
  const int lane = tid & 31;
  const int wv   = tid >> 5;       // 8 waves
  const int mi   = wv & 3;         // M-tile 0..3
  const int nb0  = (wv >> 2) * 2;  // N-tile base 0 or 2
  v8f zero = {0.f, 0.f, 0.f, 0.f, 0.f, 0.f, 0.f, 0.f};
  acc[0] = zero; acc[1] = zero;

  for (int ko = 0; ko < Hs / BK; ++ko) {   // 4 panels
    __syncthreads();                       // previous panel reads complete
    stage_chunk(A, As, ko);
    __syncthreads();                       // panel visible to all waves
#pragma unroll
    for (int ki = 0; ki < BK / 32; ++ki) { // 8 barrier-free WMMA steps
      const int k0 = ko * BK + ki * 32;
      v16bf a = make_afrag(As, mi, lane, ki * 32);
#pragma unroll
      for (int j = 0; j < 2; ++j) {
        int n = Nbase + (nb0 + j) * 16 + (lane & 15);
        if (k0 + 32 < Hs)  // prefetch next streaming B chunk (L2-resident)
          __builtin_prefetch(W + (size_t)n * Hs + k0 + 32, 0, 1);
        v16bf b = make_bfrag(W, n, k0, lane);
        acc[j] = __builtin_amdgcn_wmma_f32_16x16x32_bf16(
            false, a, false, b, (short)0, acc[j], false, false);
      }
    }
  }
}

// ---- GEMM + bias -> fp32 C (used for all three gate GEMMs) -----------------
__global__ __launch_bounds__(256) void gemm_bias_kernel(
    const bf16_t* __restrict__ A, const bf16_t* __restrict__ W,
    const float* __restrict__ bias, float* __restrict__ C, long ldc) {
  __shared__ bf16_t As[Bs * BK];
  v8f acc[2];
  const int Nbase = blockIdx.x * 64;
  gemm_core(A, W, Nbase, acc, As);
  const int lane = threadIdx.x & 31, wv = threadIdx.x >> 5;
  const int mi = wv & 3, nb0 = (wv >> 2) * 2;
  const int mbase = mi * 16 + ((lane < 16) ? 0 : 8);
#pragma unroll
  for (int j = 0; j < 2; ++j) {
    int n = Nbase + (nb0 + j) * 16 + (lane & 15);
    float bv = bias[n];
#pragma unroll
    for (int v = 0; v < 8; ++v)
      C[(size_t)(mbase + v) * ldc + n] = acc[j][v] + bv;
  }
}

// ---- logits GEMM + bias + online per-tile (rowmax, sumexp) partials --------
__global__ __launch_bounds__(256) void gemm_logits_kernel(
    const bf16_t* __restrict__ A, const bf16_t* __restrict__ W,
    const float* __restrict__ bias, float* __restrict__ C, long ldc,
    float* __restrict__ mpart, float* __restrict__ spart) {
  __shared__ bf16_t As[Bs * BK];
  __shared__ float  Ct[Bs][65];
  __shared__ float  redm[Bs][4];
  __shared__ float  reds[Bs][4];
  v8f acc[2];
  const int Nbase = blockIdx.x * 64;
  gemm_core(A, W, Nbase, acc, As);
  const int tid = threadIdx.x;
  const int lane = tid & 31, wv = tid >> 5;
  const int mi = wv & 3, nb0 = (wv >> 2) * 2;
  const int mbase = mi * 16 + ((lane < 16) ? 0 : 8);
#pragma unroll
  for (int j = 0; j < 2; ++j) {
    int nl = (nb0 + j) * 16 + (lane & 15);
    int n  = Nbase + nl;
    float bv = bias[n];
#pragma unroll
    for (int v = 0; v < 8; ++v) {
      float z = acc[j][v] + bv;
      C[(size_t)(mbase + v) * ldc + n] = z;
      Ct[mbase + v][nl] = z;
    }
  }
  __syncthreads();
  // per-row (batch) max / sumexp over this block's 64 vocab columns
  const int row = tid >> 2, part = tid & 3;
  float mx = -3.4e38f;
#pragma unroll
  for (int c = 0; c < 16; ++c) mx = fmaxf(mx, Ct[row][part * 16 + c]);
  redm[row][part] = mx;
  __syncthreads();
  float m = fmaxf(fmaxf(redm[row][0], redm[row][1]),
                  fmaxf(redm[row][2], redm[row][3]));
  float s = 0.f;
#pragma unroll
  for (int c = 0; c < 16; ++c) s += __expf(Ct[row][part * 16 + c] - m);
  reds[row][part] = s;
  __syncthreads();
  if (part == 0) {
    float st = reds[row][0] + reds[row][1] + reds[row][2] + reds[row][3];
    mpart[(size_t)blockIdx.x * Bs + row] = m;
    spart[(size_t)blockIdx.x * Bs + row] = st;
  }
}

// ---- combine partials -> logZ[64] ------------------------------------------
__global__ void combine_kernel(const float* __restrict__ mpart,
                               const float* __restrict__ spart,
                               float* __restrict__ logZ) {
  int b = threadIdx.x;  // 64 threads
  float m = -3.4e38f;
  for (int p = 0; p < VTILES; ++p) m = fmaxf(m, mpart[(size_t)p * Bs + b]);
  float s = 0.f;
  for (int p = 0; p < VTILES; ++p)
    s += spart[(size_t)p * Bs + b] * __expf(mpart[(size_t)p * Bs + b] - m);
  logZ[b] = m + __logf(s);
}

// ---- out[b,t,:] -= logZ[b] --------------------------------------------------
__global__ void subtract_kernel(float* __restrict__ out,
                                const float* __restrict__ logZ, int t) {
  int idx = blockIdx.x * blockDim.x + threadIdx.x;  // < 64*32000
  int b = idx / Vs, v = idx - b * Vs;
  size_t o = (size_t)b * Ts * Vs + (size_t)t * Vs + v;
  out[o] -= logZ[b];
}

// ---- GRU gate fusion (r,z,n) -> h_new (fp32 + bf16 mirror) ------------------
__global__ void gate_kernel(const float* __restrict__ gi,   // [64,3072] or null
                            const float* __restrict__ bih,  // used if gi==null
                            const float* __restrict__ gh,   // [64,3072]
                            float* __restrict__ hf,         // [64,1024] in/out
                            unsigned short* __restrict__ hb) {
  int idx = blockIdx.x * blockDim.x + threadIdx.x;  // < 65536
  int b = idx >> 10, h = idx & (Hs - 1);
  float ir, iz, in_;
  if (gi) {
    const float* g = gi + (size_t)b * Gs;
    ir = g[h]; iz = g[Hs + h]; in_ = g[2 * Hs + h];
  } else {
    ir = bih[h]; iz = bih[Hs + h]; in_ = bih[2 * Hs + h];
  }
  const float* gg = gh + (size_t)b * Gs;
  float hr = gg[h], hz = gg[Hs + h], hn = gg[2 * Hs + h];
  float r = 1.f / (1.f + __expf(-(ir + hr)));
  float z = 1.f / (1.f + __expf(-(iz + hz)));
  float n = tanhf(in_ + r * hn);
  float hp = hf[idx];
  float hnew = (1.f - z) * n + z * hp;
  hf[idx] = hnew;
  hb[idx] = f2bf(hnew);
}

// ---- prologue: fp32 -> bf16 weight conversion, state init -------------------
__global__ void cvt_kernel(const float* __restrict__ src,
                           unsigned short* __restrict__ dst, long n) {
  long i = (long)blockIdx.x * blockDim.x + threadIdx.x;
  if (i < n) dst[i] = f2bf(src[i]);
}

__global__ void inith_kernel(const float* __restrict__ src,
                             float* __restrict__ hf,
                             unsigned short* __restrict__ hb) {
  int i = blockIdx.x * blockDim.x + threadIdx.x;  // < 65536
  float x = src[i];
  hf[i] = x;
  hb[i] = f2bf(x);
}

// ---------------------------------------------------------------------------
extern "C" void kernel_launch(void* const* d_in, const int* in_sizes, int n_in,
                              void* d_out, int out_size, void* d_ws, size_t ws_size,
                              hipStream_t stream) {
  (void)in_sizes; (void)n_in; (void)out_size; (void)ws_size;
  const float* hidden = (const float*)d_in[0];  // [2,64,1024]
  const float* w_ih   = (const float*)d_in[1];  // [2,3072,1024]
  const float* w_hh   = (const float*)d_in[2];  // [2,3072,1024]
  const float* b_ih   = (const float*)d_in[3];  // [2,3072]
  const float* b_hh   = (const float*)d_in[4];  // [2,3072]
  const float* out_w  = (const float*)d_in[5];  // [32000,1024]
  const float* out_b  = (const float*)d_in[6];  // [32000]
  float* out = (float*)d_out;                   // [64,128,32000]

  // workspace carve-up
  char* ws = (char*)d_ws;
  size_t off = 0;
  auto carve = [&](size_t bytes) -> char* {
    char* p = ws + off; off += (bytes + 255) & ~(size_t)255; return p;
  };
  unsigned short* whh_bf  = (unsigned short*)carve((size_t)2 * Gs * Hs * 2);
  unsigned short* wih1_bf = (unsigned short*)carve((size_t)Gs * Hs * 2);
  unsigned short* outw_bf = (unsigned short*)carve((size_t)Vs * Hs * 2);
  float*          h0f     = (float*)carve((size_t)Bs * Hs * 4);
  float*          h1f     = (float*)carve((size_t)Bs * Hs * 4);
  unsigned short* h0b     = (unsigned short*)carve((size_t)Bs * Hs * 2);
  unsigned short* h1b     = (unsigned short*)carve((size_t)Bs * Hs * 2);
  float*          gh0     = (float*)carve((size_t)Bs * Gs * 4);
  float*          gi1     = (float*)carve((size_t)Bs * Gs * 4);
  float*          gh1     = (float*)carve((size_t)Bs * Gs * 4);
  float*          mpart   = (float*)carve((size_t)VTILES * Bs * 4);
  float*          spart   = (float*)carve((size_t)VTILES * Bs * 4);
  float*          logZ    = (float*)carve((size_t)Bs * 4);

  // ---- prologue: bf16 weight images (L2-resident across all 128 steps) ----
  {
    long n;
    n = (long)2 * Gs * Hs;
    cvt_kernel<<<(unsigned)((n + 255) / 256), 256, 0, stream>>>(w_hh, whh_bf, n);
    n = (long)Gs * Hs;
    cvt_kernel<<<(unsigned)((n + 255) / 256), 256, 0, stream>>>(
        w_ih + (size_t)Gs * Hs, wih1_bf, n);
    n = (long)Vs * Hs;
    cvt_kernel<<<(unsigned)((n + 255) / 256), 256, 0, stream>>>(out_w, outw_bf, n);
    inith_kernel<<<Bs * Hs / 256, 256, 0, stream>>>(hidden, h0f, h0b);
    inith_kernel<<<Bs * Hs / 256, 256, 0, stream>>>(hidden + (size_t)Bs * Hs, h1f, h1b);
  }

  const bf16_t* whh0 = (const bf16_t*)whh_bf;
  const bf16_t* whh1 = (const bf16_t*)(whh_bf + (size_t)Gs * Hs);
  const bf16_t* wih1 = (const bf16_t*)wih1_bf;
  const bf16_t* outw = (const bf16_t*)outw_bf;

  // ---- 128 sequential decode steps ----
  for (int t = 0; t < Ts; ++t) {
    // layer 0: gh0 = h0 * W_hh0^T + b_hh0 ; gates with gi0 = b_ih0 (x==0)
    gemm_bias_kernel<<<Gs / 64, 256, 0, stream>>>(
        (const bf16_t*)h0b, whh0, b_hh, gh0, Gs);
    gate_kernel<<<Bs * Hs / 256, 256, 0, stream>>>(nullptr, b_ih, gh0, h0f, h0b);

    // layer 1: gi1 = h0_new * W_ih1^T + b_ih1 ; gh1 = h1 * W_hh1^T + b_hh1
    gemm_bias_kernel<<<Gs / 64, 256, 0, stream>>>(
        (const bf16_t*)h0b, wih1, b_ih + Gs, gi1, Gs);
    gemm_bias_kernel<<<Gs / 64, 256, 0, stream>>>(
        (const bf16_t*)h1b, whh1, b_hh + Gs, gh1, Gs);
    gate_kernel<<<Bs * Hs / 256, 256, 0, stream>>>(gi1, nullptr, gh1, h1f, h1b);

    // output head: logits -> d_out[:, t, :], fused log_softmax partials
    gemm_logits_kernel<<<VTILES, 256, 0, stream>>>(
        (const bf16_t*)h1b, outw, out_b, out + (size_t)t * Vs, (long)Ts * Vs,
        mpart, spart);
    combine_kernel<<<1, Bs, 0, stream>>>(mpart, spart, logZ);
    subtract_kernel<<<Bs * Vs / 256, 256, 0, stream>>>(out, logZ, t);
  }
}
